// ProtGVPModel_60017872994803
// MI455X (gfx1250) — matrix operand order repaired and checked by reading, not compiled, not run on hardware
//
#include <hip/hip_runtime.h>

#define DEV __device__ __forceinline__

typedef __attribute__((ext_vector_type(16))) __bf16 bf16x16;
typedef __attribute__((ext_vector_type(8)))  float  f32x8;

constexpr int NN = 6400;    // nodes
constexpr int NE = 192000;  // edges
constexpr float GEPS = 1e-8f;

// ---------------------------------------------------------------- WMMA core

union FragU { bf16x16 v; uint4 q[2]; };

DEV int lane_id() { return (int)(threadIdx.x & 31); }

// A fragment (16xK tile, bf16, row-major in LDS). 16-bit A layout (ISA 7.12.2):
// lanes 0-15: row M=lane, K = k0+0..7 (v0..3) and k0+16..23 (v4..7);
// lanes 16-31: same rows, K bases +8.
DEV bf16x16 frag_a(const __bf16* base, int stride, int k0) {
  int lane = lane_id();
  const __bf16* p = base + (lane & 15) * stride + k0 + ((lane >> 4) << 3);
  FragU u;
  u.q[0] = *reinterpret_cast<const uint4*>(p);
  u.q[1] = *reinterpret_cast<const uint4*>(p + 16);
  return u.v;
}

// B fragment from pre-packed global weights: per (kChunk,nTile) 32 lanes x 16 bf16.
DEV bf16x16 frag_b(const __bf16* packed, int nTiles, int kc, int nt) {
  int lane = lane_id();
  const uint4* q = reinterpret_cast<const uint4*>(
      packed + ((size_t)((kc * nTiles + nt) * 32 + lane) << 4));
  FragU u;
  u.q[0] = q[0];
  u.q[1] = q[1];
  return u.v;
}

DEV f32x8 zero8() {
  f32x8 c;
#pragma unroll
  for (int i = 0; i < 8; ++i) c[i] = 0.f;
  return c;
}

// C += A(16xK from LDS) * B(Kx16 packed)
DEV f32x8 wgemm(const __bf16* A, int aStride, int kChunks,
                const __bf16* B, int nTiles, int nt) {
  f32x8 c = zero8();
  for (int kc = 0; kc < kChunks; ++kc) {
    c = __builtin_amdgcn_wmma_f32_16x16x32_bf16(
        false, frag_a(A, aStride, kc * 32),
        false, frag_b(B, nTiles, kc, nt),
        (short)0, c, false, false);
  }
  return c;
}

// C/D f32 tile layout: lane<16 -> col=lane rows r..; lane>=16 -> rows r+8.
DEV void store_tile(__bf16* dst, int stride, int col0, const f32x8& c) {
  int lane = lane_id();
  int col = col0 + (lane & 15);
  int rb = (lane >> 4) << 3;
#pragma unroll
  for (int r = 0; r < 8; ++r) dst[(rb + r) * stride + col] = (__bf16)c[r];
}

// vh = v @ wh for 3 channels; write vh planes and vn=sqrt(clip(sum_c vh^2)) into A.
DEV void vh_vn_tile(const __bf16* V, int vPlaneW, int vStride, int kChunks,
                    const __bf16* Bwh, int nTiles, int nt,
                    __bf16* VH, int vhPlaneW, int vhStride,
                    __bf16* A, int aStride, int vnCol0) {
  f32x8 c0 = wgemm(V, vStride, kChunks, Bwh, nTiles, nt);
  f32x8 c1 = wgemm(V + vPlaneW, vStride, kChunks, Bwh, nTiles, nt);
  f32x8 c2 = wgemm(V + 2 * vPlaneW, vStride, kChunks, Bwh, nTiles, nt);
  store_tile(VH, vhStride, nt * 16, c0);
  store_tile(VH + vhPlaneW, vhStride, nt * 16, c1);
  store_tile(VH + 2 * vhPlaneW, vhStride, nt * 16, c2);
  int lane = lane_id();
  int col = vnCol0 + nt * 16 + (lane & 15);
  int rb = (lane >> 4) << 3;
#pragma unroll
  for (int r = 0; r < 8; ++r) {
    float nn = c0[r] * c0[r] + c1[r] * c1[r] + c2[r] * c2[r];
    A[(rb + r) * aStride + col] = (__bf16)sqrtf(fmaxf(nn, GEPS));
  }
}

// v_out = vh @ wv (3 channels), optional sigmoid(|v|) gate, store to planes.
DEV void vout_tile(const __bf16* VH, int vhPlaneW, int vhStride, int kChunks,
                   const __bf16* Bwv, int nTiles, int nt, bool gate,
                   __bf16* Vd, int vdPlaneW, int vdStride) {
  f32x8 c0 = wgemm(VH, vhStride, kChunks, Bwv, nTiles, nt);
  f32x8 c1 = wgemm(VH + vhPlaneW, vhStride, kChunks, Bwv, nTiles, nt);
  f32x8 c2 = wgemm(VH + 2 * vhPlaneW, vhStride, kChunks, Bwv, nTiles, nt);
  if (gate) {
#pragma unroll
    for (int r = 0; r < 8; ++r) {
      float nn = c0[r] * c0[r] + c1[r] * c1[r] + c2[r] * c2[r];
      float nrm = sqrtf(fmaxf(nn, GEPS));
      float g = 1.f / (1.f + __expf(-nrm));
      c0[r] *= g; c1[r] *= g; c2[r] *= g;
    }
  }
  store_tile(Vd, vdStride, nt * 16, c0);
  store_tile(Vd + vdPlaneW, vdStride, nt * 16, c1);
  store_tile(Vd + 2 * vdPlaneW, vdStride, nt * 16, c2);
}

// v_out (no gate) scatter-accumulated into node accumulator (N,3,64).
DEV void vout_atomic_tile(const __bf16* VH, int vhPlaneW, int vhStride, int kChunks,
                          const __bf16* Bwv, int nTiles, int nt,
                          float* vAcc, const int* dstIds) {
  f32x8 c0 = wgemm(VH, vhStride, kChunks, Bwv, nTiles, nt);
  f32x8 c1 = wgemm(VH + vhPlaneW, vhStride, kChunks, Bwv, nTiles, nt);
  f32x8 c2 = wgemm(VH + 2 * vhPlaneW, vhStride, kChunks, Bwv, nTiles, nt);
  int lane = lane_id();
  int col = nt * 16 + (lane & 15);
  int rb = (lane >> 4) << 3;
#pragma unroll
  for (int r = 0; r < 8; ++r) {
    int n = dstIds[rb + r];
    atomicAdd(&vAcc[n * 192 + col], c0[r]);
    atomicAdd(&vAcc[n * 192 + 64 + col], c1[r]);
    atomicAdd(&vAcc[n * 192 + 128 + col], c2[r]);
  }
}

// s_out = A @ ws + bs (optional relu) -> LDS staging
DEV void sout_tile(const __bf16* A, int aStride, int kChunks,
                   const __bf16* Bws, int nTiles, int nt,
                   const float* bias, bool relu, __bf16* Sd, int sdStride) {
  f32x8 c = wgemm(A, aStride, kChunks, Bws, nTiles, nt);
  int lane = lane_id();
  int col = nt * 16 + (lane & 15);
  int rb = (lane >> 4) << 3;
  float b = bias[col];
#pragma unroll
  for (int r = 0; r < 8; ++r) {
    float v = c[r] + b;
    if (relu) v = fmaxf(v, 0.f);
    Sd[(rb + r) * sdStride + col] = (__bf16)v;
  }
}

DEV void sout_atomic_tile(const __bf16* A, int aStride, int kChunks,
                          const __bf16* Bws, int nTiles, int nt,
                          const float* bias, float* sAcc, const int* dstIds) {
  f32x8 c = wgemm(A, aStride, kChunks, Bws, nTiles, nt);
  int lane = lane_id();
  int col = nt * 16 + (lane & 15);
  int rb = (lane >> 4) << 3;
  float b = bias[col];
#pragma unroll
  for (int r = 0; r < 8; ++r) {
    int n = dstIds[rb + r];
    atomicAdd(&sAcc[n * 128 + col], c[r] + b);
  }
}

DEV float wave_sum(float x) {
  for (int o = 16; o > 0; o >>= 1) x += __shfl_xor(x, o, 32);
  return x;
}

// ---------------------------------------------------------------- weight packing

__global__ void pack_b_kernel(const float* __restrict__ W, int K, int N,
                              int Kpad, int Npad, __bf16* __restrict__ out) {
  int idx = blockIdx.x * 256 + threadIdx.x;
  int total = (Kpad >> 5) * (Npad >> 4) * 512;
  if (idx >= total) return;
  int e = idx & 15;
  int lane = (idx >> 4) & 31;
  int t = idx >> 9;
  int nT = Npad >> 4;
  int nt = t % nT, kc = t / nT;
  int k = kc * 32 + ((lane >> 4) << 3) + (e & 7) + ((e >> 3) << 4);
  int n = nt * 16 + (lane & 15);
  float v = (k < K && n < N) ? W[(size_t)k * N + n] : 0.f;
  out[idx] = (__bf16)v;
}

// ---------------------------------------------------------------- small VALU kernels

__global__ void deg_kernel(const int* __restrict__ eidx, float* __restrict__ deg) {
  int e = blockIdx.x * 256 + threadIdx.x;
  if (e < NE) atomicAdd(&deg[eidx[NE + e]], 1.f);
}

// input node embed: wv_ln + gvp(wv) (no acts). s:(N,6)->(N,128), v:(N,3,3)->(N,3ch,64)
__global__ __launch_bounds__(256)
void node_embed_kernel(const float* __restrict__ ns, const float* __restrict__ nv,
                       const float* __restrict__ g6, const float* __restrict__ b6,
                       const float* __restrict__ wh, const float* __restrict__ ws,
                       const float* __restrict__ bs, const float* __restrict__ wv,
                       float* __restrict__ sOut, float* __restrict__ vOut) {
  __shared__ float aBuf[8][80];
  __shared__ float vhBuf[8][192];
  int wid = threadIdx.x >> 5, lane = threadIdx.x & 31;
  int n = blockIdx.x * 8 + wid;
  float sl[6], mu = 0.f;
  for (int k = 0; k < 6; ++k) { sl[k] = ns[n * 6 + k]; mu += sl[k]; }
  mu /= 6.f;
  float var = 0.f;
  for (int k = 0; k < 6; ++k) { float d = sl[k] - mu; var += d * d; }
  var /= 6.f;
  float rstd = rsqrtf(var + 1e-5f);
  if (lane < 6) aBuf[wid][lane] = (sl[lane] - mu) * rstd * g6[lane] + b6[lane];
  float vv[3][3], nsum = 0.f;
  for (int i = 0; i < 3; ++i) {
    float nn = 0.f;
    for (int c = 0; c < 3; ++c) {
      float x = nv[n * 9 + i * 3 + c];
      vv[i][c] = x; nn += x * x;
    }
    nsum += fmaxf(nn, GEPS);
  }
  float ir = rsqrtf(nsum / 3.f);
  for (int j = 0; j < 2; ++j) {
    int h = lane + 32 * j;
    float a0 = 0.f, a1 = 0.f, a2 = 0.f;
    for (int i = 0; i < 3; ++i) {
      float w = wh[i * 64 + h];
      a0 += vv[i][0] * w; a1 += vv[i][1] * w; a2 += vv[i][2] * w;
    }
    a0 *= ir; a1 *= ir; a2 *= ir;
    vhBuf[wid][h] = a0; vhBuf[wid][64 + h] = a1; vhBuf[wid][128 + h] = a2;
    aBuf[wid][6 + h] = sqrtf(fmaxf(a0 * a0 + a1 * a1 + a2 * a2, GEPS));
  }
  __syncthreads();
  for (int k = 0; k < 4; ++k) {
    int o = lane + 32 * k;
    float acc = bs[o];
    for (int t = 0; t < 70; ++t) acc += aBuf[wid][t] * ws[t * 128 + o];
    sOut[n * 128 + o] = acc;
  }
  for (int j = 0; j < 2; ++j) {
    int o = lane + 32 * j;
    for (int c = 0; c < 3; ++c) {
      float acc = 0.f;
      for (int h = 0; h < 64; ++h) acc += vhBuf[wid][c * 64 + h] * wv[h * 64 + o];
      vOut[n * 192 + c * 64 + o] = acc;
    }
  }
}

// edge embed: we_ln + gvp(we). s:(E,39)->(E,32), v:(E,1,3)->(E,3)
__global__ __launch_bounds__(256)
void edge_embed_kernel(const float* __restrict__ es, const float* __restrict__ ev,
                       const float* __restrict__ g39, const float* __restrict__ b39,
                       const float* __restrict__ wh, const float* __restrict__ ws,
                       const float* __restrict__ bs, const float* __restrict__ wv,
                       float* __restrict__ esOut, float* __restrict__ evOut) {
  int e = blockIdx.x * 256 + threadIdx.x;
  if (e >= NE) return;
  float s[39], mu = 0.f;
  for (int k = 0; k < 39; ++k) { s[k] = es[e * 39 + k]; mu += s[k]; }
  mu /= 39.f;
  float var = 0.f;
  for (int k = 0; k < 39; ++k) { float d = s[k] - mu; var += d * d; }
  var /= 39.f;
  float rstd = rsqrtf(var + 1e-5f);
  for (int k = 0; k < 39; ++k) s[k] = (s[k] - mu) * rstd * g39[k] + b39[k];
  float v[3], nn = 0.f;
  for (int c = 0; c < 3; ++c) { v[c] = ev[e * 3 + c]; nn += v[c] * v[c]; }
  float ir = rsqrtf(fmaxf(nn, GEPS));
  float whs = wh[0], wvs = wv[0];
  float vh0 = v[0] * ir * whs, vh1 = v[1] * ir * whs, vh2 = v[2] * ir * whs;
  float vn = sqrtf(fmaxf(vh0 * vh0 + vh1 * vh1 + vh2 * vh2, GEPS));
  for (int o = 0; o < 32; ++o) {
    float acc = bs[o] + vn * ws[39 * 32 + o];
    for (int k = 0; k < 39; ++k) acc += s[k] * ws[k * 32 + o];
    esOut[e * 32 + o] = acc;
  }
  evOut[e * 3 + 0] = vh0 * wvs;
  evOut[e * 3 + 1] = vh1 * wvs;
  evOut[e * 3 + 2] = vh2 * wvs;
}

// residual (+ optional mean-aggregate) + GVP LayerNorm; wave per node
__global__ __launch_bounds__(256)
void node_norm_kernel(const float* __restrict__ sIn, const float* __restrict__ vIn,
                      const float* __restrict__ sAdd, const float* __restrict__ vAdd,
                      const float* __restrict__ deg,
                      const float* __restrict__ gamma, const float* __restrict__ beta,
                      float* __restrict__ sOut, float* __restrict__ vOut) {
  int wid = threadIdx.x >> 5, lane = threadIdx.x & 31;
  int n = blockIdx.x * 8 + wid;
  float inv = deg ? (1.f / fmaxf(deg[n], 1.f)) : 0.f;
  float t[4], sum = 0.f;
  for (int k = 0; k < 4; ++k) {
    int c = lane + 32 * k;
    t[k] = sIn[n * 128 + c] + (sAdd ? sAdd[n * 128 + c] * inv : 0.f);
    sum += t[k];
  }
  float mu = wave_sum(sum) / 128.f;
  float vs = 0.f;
  for (int k = 0; k < 4; ++k) { float d = t[k] - mu; vs += d * d; }
  float rstd = rsqrtf(wave_sum(vs) / 128.f + 1e-5f);
  for (int k = 0; k < 4; ++k) {
    int c = lane + 32 * k;
    sOut[n * 128 + c] = (t[k] - mu) * rstd * gamma[c] + beta[c];
  }
  float tv[2][3], nnsum = 0.f;
  for (int j = 0; j < 2; ++j) {
    int i = lane + 32 * j;
    float nn = 0.f;
    for (int c = 0; c < 3; ++c) {
      float x = vIn[n * 192 + c * 64 + i] + (vAdd ? vAdd[n * 192 + c * 64 + i] * inv : 0.f);
      tv[j][c] = x; nn += x * x;
    }
    nnsum += fmaxf(nn, GEPS);
  }
  float irr = rsqrtf(wave_sum(nnsum) / 64.f);
  for (int j = 0; j < 2; ++j) {
    int i = lane + 32 * j;
    for (int c = 0; c < 3; ++c) vOut[n * 192 + c * 64 + i] = tv[j][c] * irr;
  }
}

// final: wo_ln + gvp(wo) (relu scalar only) -> d_out (identity pyg_split for this batch)
__global__ __launch_bounds__(256)
void out_kernel(const float* __restrict__ sN, const float* __restrict__ vN,
                const float* __restrict__ g128, const float* __restrict__ b128,
                const float* __restrict__ wh, const float* __restrict__ ws,
                const float* __restrict__ bs, float* __restrict__ out) {
  __shared__ float aBuf[8][192];
  int wid = threadIdx.x >> 5, lane = threadIdx.x & 31;
  int n = blockIdx.x * 8 + wid;
  float t[4], sum = 0.f;
  for (int k = 0; k < 4; ++k) { int c = lane + 32 * k; t[k] = sN[n * 128 + c]; sum += t[k]; }
  float mu = wave_sum(sum) / 128.f;
  float vs = 0.f;
  for (int k = 0; k < 4; ++k) { float d = t[k] - mu; vs += d * d; }
  float rstd = rsqrtf(wave_sum(vs) / 128.f + 1e-5f);
  for (int k = 0; k < 4; ++k) {
    int c = lane + 32 * k;
    aBuf[wid][c] = (t[k] - mu) * rstd * g128[c] + b128[c];
  }
  float nnsum = 0.f;
  for (int j = 0; j < 2; ++j) {
    int i = lane + 32 * j;
    float nn = 0.f;
    for (int c = 0; c < 3; ++c) { float x = vN[n * 192 + c * 64 + i]; nn += x * x; }
    nnsum += fmaxf(nn, GEPS);
  }
  float ir = rsqrtf(wave_sum(nnsum) / 64.f);
  for (int j = 0; j < 2; ++j) {
    int h = lane + 32 * j;
    float a0 = 0.f, a1 = 0.f, a2 = 0.f;
    for (int i = 0; i < 64; ++i) {
      float w = wh[i * 64 + h];
      a0 += vN[n * 192 + i] * w;
      a1 += vN[n * 192 + 64 + i] * w;
      a2 += vN[n * 192 + 128 + i] * w;
    }
    a0 *= ir; a1 *= ir; a2 *= ir;
    aBuf[wid][128 + h] = sqrtf(fmaxf(a0 * a0 + a1 * a1 + a2 * a2, GEPS));
  }
  __syncthreads();
  for (int k = 0; k < 4; ++k) {
    int o = lane + 32 * k;
    float acc = bs[o];
    for (int q = 0; q < 192; ++q) acc += aBuf[wid][q] * ws[q * 128 + o];
    out[n * 128 + o] = fmaxf(acc, 0.f);
  }
}

// ---------------------------------------------------------------- fused WMMA kernels

struct ConvW {
  const __bf16 *wh0, *ws0, *wv0, *wh1, *ws1, *wv1, *wh2, *ws2, *wv2;
  const float *bs0, *bs1, *bs2;
};

// one wave per 16-edge tile; m0 -> m1 -> m2, atomic scatter of messages
__global__ __launch_bounds__(32)
void edge_msg_kernel(const float* __restrict__ sN, const float* __restrict__ vN,
                     const float* __restrict__ esF, const float* __restrict__ evF,
                     const int* __restrict__ eidx, ConvW W,
                     float* __restrict__ sAcc, float* __restrict__ vAcc) {
  __shared__ __align__(16) __bf16 sA[16 * 448];   // [ms(417)|pad] then [s|vn]
  __shared__ __align__(16) __bf16 sMV[16 * 480];  // 3 planes x160: v input
  __shared__ __align__(16) __bf16 sVH[16 * 480];  // 3 planes x160: vh
  __shared__ __align__(16) __bf16 sS[16 * 128];   // s staging
  __shared__ int eSrc[16], eDst[16];
  const int lane = threadIdx.x;
  const int e0 = blockIdx.x * 16;

  { unsigned int* z;
    z = (unsigned int*)sA;  for (int t = lane; t < 16 * 448 / 2; t += 32) z[t] = 0u;
    z = (unsigned int*)sMV; for (int t = lane; t < 16 * 480 / 2; t += 32) z[t] = 0u;
    z = (unsigned int*)sVH; for (int t = lane; t < 16 * 480 / 2; t += 32) z[t] = 0u; }
  if (lane < 16) { eSrc[lane] = eidx[e0 + lane]; eDst[lane] = eidx[NE + e0 + lane]; }
  __syncthreads();

  // stage ms = [s_src | es | s_dst] and per-channel mv = [v_src | ev | v_dst]
  for (int t = lane; t < 16 * 128; t += 32) { int r = t >> 7, c = t & 127; sA[r * 448 + c] = (__bf16)sN[eSrc[r] * 128 + c]; }
  for (int t = lane; t < 16 * 32; t += 32)  { int r = t >> 5, c = t & 31;  sA[r * 448 + 128 + c] = (__bf16)esF[(e0 + r) * 32 + c]; }
  for (int t = lane; t < 16 * 128; t += 32) { int r = t >> 7, c = t & 127; sA[r * 448 + 160 + c] = (__bf16)sN[eDst[r] * 128 + c]; }
  for (int cc = 0; cc < 3; ++cc) {
    for (int t = lane; t < 16 * 64; t += 32) { int r = t >> 6, i = t & 63; sMV[r * 480 + cc * 160 + i] = (__bf16)vN[eSrc[r] * 192 + cc * 64 + i]; }
    for (int t = lane; t < 16 * 64; t += 32) { int r = t >> 6, i = t & 63; sMV[r * 480 + cc * 160 + 65 + i] = (__bf16)vN[eDst[r] * 192 + cc * 64 + i]; }
    if (lane < 16) sMV[lane * 480 + cc * 160 + 64] = (__bf16)evF[(e0 + lane) * 3 + cc];
  }
  __syncthreads();

  // ---- m0: vi=129 (Kpad160) h=129 (Npad144) si=288 so=128 vo=64, relu+gate
  for (int nt = 0; nt < 9; ++nt)
    vh_vn_tile(sMV, 160, 480, 5, W.wh0, 9, nt, sVH, 160, 480, sA, 448, 288);
  for (int nt = 0; nt < 8; ++nt)
    sout_tile(sA, 448, 14, W.ws0, 8, nt, W.bs0, true, sS, 128);
  for (int nt = 0; nt < 4; ++nt)
    vout_tile(sVH, 160, 480, 5, W.wv0, 4, nt, true, sMV, 160, 480);
  __syncthreads();
  for (int t = lane; t < 16 * 128; t += 32) { int r = t >> 7, c = t & 127; sA[r * 448 + c] = sS[r * 128 + c]; }
  __syncthreads();

  // ---- m1: vi=64 h=64 si=128 so=128 vo=64, relu+gate
  for (int nt = 0; nt < 4; ++nt)
    vh_vn_tile(sMV, 160, 480, 2, W.wh1, 4, nt, sVH, 160, 480, sA, 448, 128);
  for (int nt = 0; nt < 8; ++nt)
    sout_tile(sA, 448, 6, W.ws1, 8, nt, W.bs1, true, sS, 128);
  for (int nt = 0; nt < 4; ++nt)
    vout_tile(sVH, 160, 480, 2, W.wv1, 4, nt, true, sMV, 160, 480);
  __syncthreads();
  for (int t = lane; t < 16 * 128; t += 32) { int r = t >> 7, c = t & 127; sA[r * 448 + c] = sS[r * 128 + c]; }
  __syncthreads();

  // ---- m2: no acts, atomic scatter-add of messages
  for (int nt = 0; nt < 4; ++nt)
    vh_vn_tile(sMV, 160, 480, 2, W.wh2, 4, nt, sVH, 160, 480, sA, 448, 128);
  for (int nt = 0; nt < 8; ++nt)
    sout_atomic_tile(sA, 448, 6, W.ws2, 8, nt, W.bs2, sAcc, eDst);
  for (int nt = 0; nt < 4; ++nt)
    vout_atomic_tile(sVH, 160, 480, 2, W.wv2, 4, nt, vAcc, eDst);
}

struct FFW {
  const __bf16 *wh0, *ws0, *wv0, *wh1, *ws1, *wv1;
  const float *bs0, *bs1;
};

// one wave per 16-node tile; ff0 -> ff1, writes (sMid+fs), (vMid+fv)
__global__ __launch_bounds__(32)
void node_ff_kernel(const float* __restrict__ sMid, const float* __restrict__ vMid,
                    FFW W, float* __restrict__ sSum, float* __restrict__ vSum) {
  __shared__ __align__(16) __bf16 sA[16 * 640];
  __shared__ __align__(16) __bf16 sMV[16 * 384];  // 3 planes x128
  __shared__ __align__(16) __bf16 sVH[16 * 384];
  __shared__ __align__(16) __bf16 sS[16 * 512];
  const int lane = threadIdx.x;
  const int n0 = blockIdx.x * 16;

  { unsigned int* z;
    z = (unsigned int*)sA;  for (int t = lane; t < 16 * 640 / 2; t += 32) z[t] = 0u;
    z = (unsigned int*)sMV; for (int t = lane; t < 16 * 384 / 2; t += 32) z[t] = 0u;
    z = (unsigned int*)sVH; for (int t = lane; t < 16 * 384 / 2; t += 32) z[t] = 0u; }
  __syncthreads();
  for (int t = lane; t < 16 * 128; t += 32) { int r = t >> 7, c = t & 127; sA[r * 640 + c] = (__bf16)sMid[(n0 + r) * 128 + c]; }
  for (int cc = 0; cc < 3; ++cc)
    for (int t = lane; t < 16 * 64; t += 32) { int r = t >> 6, i = t & 63; sMV[r * 384 + cc * 128 + i] = (__bf16)vMid[(n0 + r) * 192 + cc * 64 + i]; }
  __syncthreads();

  // ---- ff0: vi=64 h=128 si=128 so=512 vo=128, relu+gate
  for (int nt = 0; nt < 8; ++nt)
    vh_vn_tile(sMV, 128, 384, 2, W.wh0, 8, nt, sVH, 128, 384, sA, 640, 128);
  for (int nt = 0; nt < 32; ++nt)
    sout_tile(sA, 640, 8, W.ws0, 32, nt, W.bs0, true, sS, 512);
  for (int nt = 0; nt < 8; ++nt)
    vout_tile(sVH, 128, 384, 4, W.wv0, 8, nt, true, sMV, 128, 384);
  __syncthreads();
  for (int t = lane; t < 16 * 512; t += 32) { int r = t >> 9, c = t & 511; sA[r * 640 + c] = sS[r * 512 + c]; }
  __syncthreads();

  // ---- ff1: vi=128 h=128 si=512 so=128 vo=64, no acts; add residual
  for (int nt = 0; nt < 8; ++nt)
    vh_vn_tile(sMV, 128, 384, 4, W.wh1, 8, nt, sVH, 128, 384, sA, 640, 512);
  for (int nt = 0; nt < 8; ++nt) {
    f32x8 c = wgemm(sA, 640, 20, W.ws1, 8, nt);
    int col = nt * 16 + (lane & 15);
    int rb = (lane >> 4) << 3;
    float b = W.bs1[col];
#pragma unroll
    for (int r = 0; r < 8; ++r) {
      int n = n0 + rb + r;
      sSum[n * 128 + col] = sMid[n * 128 + col] + c[r] + b;
    }
  }
  for (int nt = 0; nt < 4; ++nt) {
    f32x8 c0 = wgemm(sVH, 384, 4, W.wv1, 4, nt);
    f32x8 c1 = wgemm(sVH + 128, 384, 4, W.wv1, 4, nt);
    f32x8 c2 = wgemm(sVH + 256, 384, 4, W.wv1, 4, nt);
    int col = nt * 16 + (lane & 15);
    int rb = (lane >> 4) << 3;
#pragma unroll
    for (int r = 0; r < 8; ++r) {
      int n = n0 + rb + r;
      vSum[n * 192 + col]       = vMid[n * 192 + col] + c0[r];
      vSum[n * 192 + 64 + col]  = vMid[n * 192 + 64 + col] + c1[r];
      vSum[n * 192 + 128 + col] = vMid[n * 192 + 128 + col] + c2[r];
    }
  }
}

// ---------------------------------------------------------------- host side

extern "C" void kernel_launch(void* const* d_in, const int* in_sizes, int n_in,
                              void* d_out, int out_size, void* d_ws, size_t ws_size,
                              hipStream_t stream) {
  (void)in_sizes; (void)out_size; (void)ws_size;
  if (n_in < 97) return;
  auto F = [&](int i) { return (const float*)d_in[i]; };

  const float* node_s = F(0);
  const float* node_v = F(1);
  const float* edge_s = F(2);
  const float* edge_v = F(3);
  const int* eidx = (const int*)d_in[4];

  struct G4 { const float *bs, *wh, *ws, *wv; };
  G4 m0[3], m1[3], m2[3], f0[3], f1[3];
  const float *ln0b[3], *ln0g[3], *ln1b[3], *ln1g[3];
  int idx = 8;  // params leaves in jax pytree order (sorted keys, DFS)
  for (int l = 0; l < 3; ++l) {
    m0[l] = {F(idx), F(idx + 1), F(idx + 2), F(idx + 3)}; idx += 4;
    m1[l] = {F(idx), F(idx + 1), F(idx + 2), F(idx + 3)}; idx += 4;
    m2[l] = {F(idx), F(idx + 1), F(idx + 2), F(idx + 3)}; idx += 4;
    f0[l] = {F(idx), F(idx + 1), F(idx + 2), F(idx + 3)}; idx += 4;
    f1[l] = {F(idx), F(idx + 1), F(idx + 2), F(idx + 3)}; idx += 4;
    ln0b[l] = F(idx); ln0g[l] = F(idx + 1); idx += 2;
    ln1b[l] = F(idx); ln1g[l] = F(idx + 1); idx += 2;
  }
  G4 we = {F(idx), F(idx + 1), F(idx + 2), F(idx + 3)}; idx += 4;
  const float *welnB = F(idx), *welnG = F(idx + 1); idx += 2;
  const float *woBs = F(idx), *woWh = F(idx + 1), *woWs = F(idx + 2); idx += 3;
  const float *wolnB = F(idx), *wolnG = F(idx + 1); idx += 2;
  G4 wvp = {F(idx), F(idx + 1), F(idx + 2), F(idx + 3)}; idx += 4;
  const float *wvlnB = F(idx), *wvlnG = F(idx + 1); idx += 2;

  // workspace carve-out
  size_t off = 0;
  auto alloc = [&](size_t bytes) {
    off = (off + 255) & ~(size_t)255;
    void* p = (char*)d_ws + off;
    off += bytes;
    return p;
  };
  float* sCur = (float*)alloc((size_t)NN * 128 * 4);
  float* vCur = (float*)alloc((size_t)NN * 192 * 4);
  float* sMid = (float*)alloc((size_t)NN * 128 * 4);
  float* vMid = (float*)alloc((size_t)NN * 192 * 4);
  float* sAcc = (float*)alloc((size_t)NN * 128 * 4);
  float* vAcc = (float*)alloc((size_t)NN * 192 * 4);
  float* esF  = (float*)alloc((size_t)NE * 32 * 4);
  float* evF  = (float*)alloc((size_t)NE * 3 * 4);
  float* deg  = (float*)alloc((size_t)NN * 4);

  auto packB = [&](const float* W, int K, int N, int Kpad, int Npad) {
    __bf16* p = (__bf16*)alloc((size_t)Kpad * Npad * 2);
    int total = (Kpad / 32) * (Npad / 16) * 512;
    pack_b_kernel<<<(total + 255) / 256, 256, 0, stream>>>(W, K, N, Kpad, Npad, p);
    return (const __bf16*)p;
  };

  ConvW cw[3]; FFW fw[3];
  for (int l = 0; l < 3; ++l) {
    cw[l].wh0 = packB(m0[l].wh, 129, 129, 160, 144);
    cw[l].ws0 = packB(m0[l].ws, 417, 128, 448, 128);
    cw[l].wv0 = packB(m0[l].wv, 129, 64, 160, 64);
    cw[l].wh1 = packB(m1[l].wh, 64, 64, 64, 64);
    cw[l].ws1 = packB(m1[l].ws, 192, 128, 192, 128);
    cw[l].wv1 = packB(m1[l].wv, 64, 64, 64, 64);
    cw[l].wh2 = packB(m2[l].wh, 64, 64, 64, 64);
    cw[l].ws2 = packB(m2[l].ws, 192, 128, 192, 128);
    cw[l].wv2 = packB(m2[l].wv, 64, 64, 64, 64);
    cw[l].bs0 = m0[l].bs; cw[l].bs1 = m1[l].bs; cw[l].bs2 = m2[l].bs;
    fw[l].wh0 = packB(f0[l].wh, 64, 128, 64, 128);
    fw[l].ws0 = packB(f0[l].ws, 256, 512, 256, 512);
    fw[l].wv0 = packB(f0[l].wv, 128, 128, 128, 128);
    fw[l].wh1 = packB(f1[l].wh, 128, 128, 128, 128);
    fw[l].ws1 = packB(f1[l].ws, 640, 128, 640, 128);
    fw[l].wv1 = packB(f1[l].wv, 128, 64, 128, 64);
    fw[l].bs0 = f0[l].bs; fw[l].bs1 = f1[l].bs;
  }

  hipMemsetAsync(deg, 0, (size_t)NN * 4, stream);
  deg_kernel<<<(NE + 255) / 256, 256, 0, stream>>>(eidx, deg);
  node_embed_kernel<<<NN / 8, 256, 0, stream>>>(node_s, node_v, wvlnG, wvlnB,
                                                wvp.wh, wvp.ws, wvp.bs, wvp.wv, sCur, vCur);
  edge_embed_kernel<<<(NE + 255) / 256, 256, 0, stream>>>(edge_s, edge_v, welnG, welnB,
                                                          we.wh, we.ws, we.bs, we.wv, esF, evF);
  for (int l = 0; l < 3; ++l) {
    hipMemsetAsync(sAcc, 0, (size_t)NN * 128 * 4, stream);
    hipMemsetAsync(vAcc, 0, (size_t)NN * 192 * 4, stream);
    edge_msg_kernel<<<NE / 16, 32, 0, stream>>>(sCur, vCur, esF, evF, eidx, cw[l], sAcc, vAcc);
    node_norm_kernel<<<NN / 8, 256, 0, stream>>>(sCur, vCur, sAcc, vAcc, deg,
                                                 ln0g[l], ln0b[l], sMid, vMid);
    node_ff_kernel<<<NN / 16, 32, 0, stream>>>(sMid, vMid, fw[l], sAcc, vAcc);
    node_norm_kernel<<<NN / 8, 256, 0, stream>>>(sAcc, vAcc, nullptr, nullptr, nullptr,
                                                 ln1g[l], ln1b[l], sCur, vCur);
  }
  out_kernel<<<NN / 8, 256, 0, stream>>>(sCur, vCur, wolnG, wolnB, woWh, woWs, woBs,
                                         (float*)d_out);
}
